// MultiHeadAttention_76794015252889
// MI455X (gfx1250) — compile-verified, matrix operand order checked by reading, not run on hardware
//
#include <hip/hip_runtime.h>
#include <math.h>

typedef __attribute__((ext_vector_type(16))) _Float16     v16h;
typedef __attribute__((ext_vector_type(8)))  float        v8f;
typedef __attribute__((ext_vector_type(4)))  unsigned int v4u;
typedef __attribute__((ext_vector_type(8)))  int          v8i;
typedef __attribute__((ext_vector_type(4)))  int          v4i;

#define WMMA_F16(a, b, c) \
  __builtin_amdgcn_wmma_f32_16x16x32_f16(false, (a), false, (b), (short)0, (c), false, false)

// ---------------------------------------------------------------------------
// WMMA operand loaders (ISA 7.12.2, wave32). A: lane%16 = row, lanes>=16 take
// the alternate k-octets {8..15,24..31}. B mirrored with lane%16 = column.
// ---------------------------------------------------------------------------
__device__ __forceinline__ v16h ldmat_f16(const _Float16* base, int kstride, int lane) {
  const int kh = (lane >> 4) & 1;
  v16h r;
#pragma unroll
  for (int j = 0; j < 8; ++j) r[j] = base[(size_t)(kh * 8 + j) * kstride];
#pragma unroll
  for (int j = 0; j < 8; ++j) r[j + 8] = base[(size_t)(16 + kh * 8 + j) * kstride];
  return r;
}

__device__ __forceinline__ v16h ldmat_f32(const float* base, int kstride, int lane) {
  const int kh = (lane >> 4) & 1;
  v16h r;
#pragma unroll
  for (int j = 0; j < 8; ++j) r[j] = (_Float16)base[(size_t)(kh * 8 + j) * kstride];
#pragma unroll
  for (int j = 0; j < 8; ++j) r[j + 8] = (_Float16)base[(size_t)(16 + kh * 8 + j) * kstride];
  return r;
}

// ---------------------------------------------------------------------------
// Tensor Data Mover: 2D tile (tile_d0 contiguous elems x tile_d1 rows) from
// global -> LDS. Descriptor per ISA 8.3/8.4: group0 = {count|flags, lds_addr,
// global_addr[31:0], global_addr[56:32]|type=2}; group1 packs data_size,
// tensor_dim0/1 (== tile dims; tile fully in-bounds), tile_dim0/1 and
// tensor_dim0_stride. Groups 2/3 (dims 2..4) unused -> zeros. Issued by one
// wave; completion tracked via TENSORcnt.
// ---------------------------------------------------------------------------
__device__ __forceinline__ void tdm_load_2d(const void* lds_dst, const void* gsrc,
                                            unsigned elem_log2,            // 1=2B, 2=4B
                                            unsigned tile_d0, unsigned tile_d1,
                                            unsigned long long row_stride_elems) {
  const unsigned long long ga = (unsigned long long)(uintptr_t)gsrc;
  const unsigned lds_addr = (unsigned)(uintptr_t)lds_dst;  // low 32b of flat = LDS offset
  v4u g0;
  g0[0] = 1u;                                              // count=1, user descriptor
  g0[1] = lds_addr;
  g0[2] = (unsigned)ga;
  g0[3] = (unsigned)((ga >> 32) & 0x1FFFFFFu) | (2u << 30);  // addr[56:32] | type=2
  v8i g1;
  g1[0] = (int)(elem_log2 << 16);                          // data_size; wg_mask=0
  g1[1] = (int)((tile_d0 & 0xFFFFu) << 16);                // tensor_dim0[15:0]
  g1[2] = (int)((tile_d0 >> 16) | ((tile_d1 & 0xFFFFu) << 16)); // tdim0 hi | tdim1 lo
  g1[3] = (int)((tile_d1 >> 16) | (tile_d0 << 16));        // tdim1 hi | tile_dim0
  g1[4] = (int)(tile_d1 & 0xFFFFu);                        // tile_dim1 (tile_dim2=0)
  g1[5] = (int)(row_stride_elems & 0xFFFFFFFFu);           // tensor_dim0_stride lo32
  g1[6] = (int)((row_stride_elems >> 32) & 0xFFFFu);       // stride hi16
  g1[7] = 0;
  const v4i z4 = {0, 0, 0, 0};
  const v8i z8 = {0, 0, 0, 0, 0, 0, 0, 0};
  __builtin_amdgcn_tensor_load_to_lds(g0, g1, z4, z4, z8, 0);
}

// ---------------------------------------------------------------------------
// Projection GEMM: Y[m][n] = sum_k X[m][k]*W[n][k] -> f16.
// Block tile 32(M) x 64(N), 8 waves as 2x4 of 16x16 WMMA tiles. A/B K-chunks
// (32 wide) streamed into LDS by TDM, double-buffered (compute overlaps DMA).
// ---------------------------------------------------------------------------
__global__ void proj_gemm_f16(const float* __restrict__ X, const float* __restrict__ W,
                              _Float16* __restrict__ Y, int N, int K) {
  __shared__ float sA[2][32 * 32];   // 2 x 4 KB
  __shared__ float sB[2][64 * 32];   // 2 x 8 KB

  const int t = threadIdx.x, w = t >> 5, lane = t & 31;
  const int ntiles = N >> 6;                       // 64-wide N tiles
  const int m0 = (blockIdx.x / ntiles) << 5;
  const int n0 = (blockIdx.x % ntiles) << 6;
  const int wm = w >> 2, wn = w & 3;               // wave grid 2x4
  const int cw = lane & 15, rb = (lane >> 4) * 8;

  if (w == 0) {                                    // prime chunk 0
    tdm_load_2d(sA[0], X + (size_t)m0 * K, 2u, 32u, 32u, (unsigned long long)K);
    tdm_load_2d(sB[0], W + (size_t)n0 * K, 2u, 32u, 64u, (unsigned long long)K);
  }
  v8f acc = {};
  const int nchunk = K >> 5;
  for (int c = 0; c < nchunk; ++c) {
    if (w == 0) {
      if (c + 1 < nchunk) {
        const int kb = (c + 1) << 5;
        tdm_load_2d(sA[(c + 1) & 1], X + (size_t)m0 * K + kb, 2u, 32u, 32u, (unsigned long long)K);
        tdm_load_2d(sB[(c + 1) & 1], W + (size_t)n0 * K + kb, 2u, 32u, 64u, (unsigned long long)K);
        __builtin_amdgcn_s_wait_tensorcnt(2);      // chunk c landed; c+1 in flight
      } else {
        __builtin_amdgcn_s_wait_tensorcnt(0);
      }
    }
    __syncthreads();
    const float* pa = sA[c & 1] + (wm * 16 + cw) * 32;
    const float* pb = sB[c & 1] + (wn * 16 + cw) * 32;
    acc = WMMA_F16(ldmat_f32(pa, 1, lane), ldmat_f32(pb, 1, lane), acc);
    __syncthreads();                               // release buffer c&1
  }
#pragma unroll
  for (int j = 0; j < 8; ++j)
    Y[(size_t)(m0 + wm * 16 + rb + j) * N + n0 + wn * 16 + cw] = (_Float16)acc[j];
}

// ---------------------------------------------------------------------------
// Per-(row, head) L2 normalize in place: x *= scale / max(||x||, 1e-12).
// ---------------------------------------------------------------------------
__global__ void l2norm_inplace_f16(_Float16* __restrict__ Y, float scale) {
  const int wid  = blockIdx.x * (blockDim.x >> 5) + (threadIdx.x >> 5);
  const int lane = threadIdx.x & 31;
  _Float16* p = Y + (size_t)wid * 64;
  float x0 = (float)p[lane], x1 = (float)p[lane + 32];
  float s = x0 * x0 + x1 * x1;
#pragma unroll
  for (int m = 16; m >= 1; m >>= 1) s += __shfl_xor(s, m, 32);
  const float inv = scale / fmaxf(sqrtf(s), 1e-12f);
  p[lane]      = (_Float16)(x0 * inv);
  p[lane + 32] = (_Float16)(x1 * inv);
}

// ---------------------------------------------------------------------------
// Fused attention. Block = (batch, 16-query tile), loops 16 heads.
// K / V head-slices streamed 256-keys-at-a-time into a shared 2x32KB LDS
// double buffer via TDM; V chunk 0 prefetch overlaps the softmax pass.
// Scores (16x2048 f32, 128 KB) live in LDS; attn_avg via exclusive global RMW
// (L2-resident, no atomics). Total LDS ~198 KB < 320 KB.
// ---------------------------------------------------------------------------
__global__ void attn_fused(const _Float16* __restrict__ qn, const _Float16* __restrict__ kn,
                           const _Float16* __restrict__ vn, float* __restrict__ out,
                           float* __restrict__ avg) {
  constexpr int L = 2048, D = 1024, CK = 256;      // CK keys per staged chunk
  __shared__ float    sS[16 * L];                  // 128 KB scores/probs
  __shared__ _Float16 sT[2][CK * 64];              // 2 x 32 KB K/V staging
  __shared__ _Float16 sQ[16 * 64];                 // 2 KB
  __shared__ float    sRed[16 * 64];               // 4 KB

  const int t = threadIdx.x, w = t >> 5, lane = t & 31;
  const int b  = blockIdx.x >> 7;
  const int q0 = (blockIdx.x & 127) << 4;
  const int cw = lane & 15, rb = (lane >> 4) * 8;

  for (int h = 0; h < 16; ++h) {
    {  // stage q-hat tile (16x64 f16) cooperatively
      const int idx = t * 4;
      const int r = idx >> 6, d = idx & 63;
      const _Float16* src = qn + (size_t)(b * L + q0 + r) * D + h * 64 + d;
#pragma unroll
      for (int j = 0; j < 4; ++j) sQ[idx + j] = src[j];
    }
    if (w == 0)  // prime K chunk 0
      tdm_load_2d(sT[0], kn + (size_t)(b * L) * D + h * 64, 1u, 64u, CK, (unsigned long long)D);
    __syncthreads();

    // ---- S = q-hat @ k-hat^T, K streamed in 8 chunks of 256 keys ----
    for (int c = 0; c < 8; ++c) {
      if (w == 0) {
        if (c < 7) {
          tdm_load_2d(sT[(c + 1) & 1], kn + (size_t)(b * L + (c + 1) * CK) * D + h * 64,
                      1u, 64u, CK, (unsigned long long)D);
          __builtin_amdgcn_s_wait_tensorcnt(1);
        } else {
          __builtin_amdgcn_s_wait_tensorcnt(0);
        }
      }
      __syncthreads();
      const _Float16* buf = sT[c & 1];
      for (int tl = w; tl < 16; tl += 8) {         // 16 key-tiles, 2 per wave
        const int key0 = c * CK + tl * 16;
        const _Float16* bbase = buf + (tl * 16 + cw) * 64;
        v8f s = {};
        s = WMMA_F16(ldmat_f16(sQ + cw * 64, 1, lane),      ldmat_f16(bbase, 1, lane),      s);
        s = WMMA_F16(ldmat_f16(sQ + cw * 64 + 32, 1, lane), ldmat_f16(bbase + 32, 1, lane), s);
#pragma unroll
        for (int j = 0; j < 8; ++j) sS[(rb + j) * L + key0 + cw] = s[j];
      }
      __syncthreads();
    }

    if (w == 0)  // prefetch V chunk 0 -- DMA overlaps the softmax below
      tdm_load_2d(sT[0], vn + (size_t)(b * L) * D + h * 64, 1u, 64u, CK, (unsigned long long)D);

    // ---- exact softmax (2 rows/wave) + attn_avg RMW fold ----
    for (int rr = w * 2; rr < w * 2 + 2; ++rr) {
      float* row = sS + rr * L;
      float mx = -INFINITY;
      for (int c = lane; c < L; c += 32) mx = fmaxf(mx, row[c]);
#pragma unroll
      for (int m = 16; m >= 1; m >>= 1) mx = fmaxf(mx, __shfl_xor(mx, m, 32));
      float sum = 0.f;
      for (int c = lane; c < L; c += 32) { float p = __expf(row[c] - mx); row[c] = p; sum += p; }
#pragma unroll
      for (int m = 16; m >= 1; m >>= 1) sum += __shfl_xor(sum, m, 32);
      const float inv = 1.f / sum;
      float* avgRow = avg + ((size_t)b * L + q0 + rr) * L;
      for (int c = lane; c < L; c += 32) {
        const float p = row[c] * inv;
        row[c] = p;
        float a = p * 0.0625f;
        if (h != 0) a += avgRow[c];
        avgRow[c] = a;
      }
    }
    __syncthreads();

    // ---- O = P @ V, V streamed in 8 chunks; wave = (key-half g, col tile ct) ----
    const int g = w >> 2, ct = w & 3;
    const int col = ct * 16 + cw;
    v8f acc = {};
    for (int c = 0; c < 8; ++c) {
      if (w == 0) {
        if (c < 7) {
          tdm_load_2d(sT[(c + 1) & 1], vn + (size_t)(b * L + (c + 1) * CK) * D + h * 64,
                      1u, 64u, CK, (unsigned long long)D);
          __builtin_amdgcn_s_wait_tensorcnt(1);
        } else {
          __builtin_amdgcn_s_wait_tensorcnt(0);
        }
      }
      __syncthreads();
      const _Float16* buf = sT[c & 1];
#pragma unroll
      for (int s32 = 0; s32 < 4; ++s32) {          // 128 keys per wave-group
        const int kin = g * 128 + s32 * 32;
        v16h a  = ldmat_f32(sS + cw * L + c * CK + kin, 1, lane);
        v16h bv = ldmat_f16(buf + kin * 64 + col, 64, lane);
        acc = WMMA_F16(a, bv, acc);
      }
      __syncthreads();
    }
    if (g == 1) {
#pragma unroll
      for (int j = 0; j < 8; ++j) sRed[(rb + j) * 64 + col] = acc[j];
    }
    __syncthreads();
    if (g == 0) {
      float* orow = out + (((size_t)b * 16 + h) * L + q0) * 64;
#pragma unroll
      for (int j = 0; j < 8; ++j)
        orow[(size_t)(rb + j) * 64 + col] = acc[j] + sRed[(rb + j) * 64 + col];
    }
    __syncthreads();
  }
}

// ---------------------------------------------------------------------------
extern "C" void kernel_launch(void* const* d_in, const int* in_sizes, int n_in,
                              void* d_out, int out_size, void* d_ws, size_t ws_size,
                              hipStream_t stream) {
  (void)in_sizes; (void)n_in; (void)out_size; (void)ws_size;
  const float* q  = (const float*)d_in[0];
  const float* k  = (const float*)d_in[1];
  const float* v  = (const float*)d_in[2];
  const float* wq = (const float*)d_in[3];
  const float* wk = (const float*)d_in[4];
  const float* wv = (const float*)d_in[5];

  constexpr int B = 2, L = 2048, D = 1024, H = 16;
  constexpr size_t ROWS = (size_t)B * L;
  _Float16* qn = (_Float16*)d_ws;                 // 8 MB each, f16 [B*L][H*64]
  _Float16* kn = qn + ROWS * D;
  _Float16* vn = kn + ROWS * D;

  float* out = (float*)d_out;                     // [B,H,L,64]
  float* avg = out + (size_t)B * H * L * 64;      // [B,L,L]

  dim3 blk(256);
  // (4096/32)*(1024/64) = 2048 blocks per projection
  proj_gemm_f16<<<2048, blk, 0, stream>>>(q, wq, qn, D, D);
  proj_gemm_f16<<<2048, blk, 0, stream>>>(k, wk, kn, D, D);
  proj_gemm_f16<<<2048, blk, 0, stream>>>(v, wv, vn, D, D);
  // B*L*H = 65536 head-vectors, 8 waves/block
  l2norm_inplace_f16<<<8192, blk, 0, stream>>>(qn, 0.125f);   // fold 1/sqrt(d_k)
  l2norm_inplace_f16<<<8192, blk, 0, stream>>>(kn, 1.0f);
  attn_fused<<<B * (L / 16), blk, 0, stream>>>(qn, kn, vn, out, avg);
}
